// MultiHeadAttention_17643725652045
// MI455X (gfx1250) — matrix operand order repaired and checked by reading, not compiled, not run on hardware
//
#include <hip/hip_runtime.h>

typedef unsigned short u16;
typedef unsigned int   u32;
typedef __attribute__((ext_vector_type(16))) __bf16 v16bf;
typedef __attribute__((ext_vector_type(8)))  float  v8f;

static constexpr int Bn = 4, Tn = 2048, Cn = 1024, Hn = 16, DHn = 64;

union Frag {
    v16bf v;
    uint4 q[2];
    u16   s[16];
    __bf16 b[16];
};

// Native f32 -> bf16 (backend selects v_cvt_pk_bf16_f32 / scalar cvt, RNE).
__device__ __forceinline__ u16 f2bf(float f) {
    union { __bf16 h; u16 u; } c;
    c.h = (__bf16)f;
    return c.u;
}

__device__ __forceinline__ float fexp2(float x) {
#if __has_builtin(__builtin_amdgcn_exp2f)
    return __builtin_amdgcn_exp2f(x);
#else
    return exp2f(x);
#endif
}

__device__ __forceinline__ float red_max16(float v) {
    v = fmaxf(v, __shfl_xor(v, 1, 32));
    v = fmaxf(v, __shfl_xor(v, 2, 32));
    v = fmaxf(v, __shfl_xor(v, 4, 32));
    v = fmaxf(v, __shfl_xor(v, 8, 32));
    return v;
}
__device__ __forceinline__ float red_sum16(float v) {
    v += __shfl_xor(v, 1, 32);
    v += __shfl_xor(v, 2, 32);
    v += __shfl_xor(v, 4, 32);
    v += __shfl_xor(v, 8, 32);
    return v;
}

// Async global -> LDS copy of one 16B chunk (ASYNCcnt-tracked).
__device__ __forceinline__ void async_b128(const u16* gsrc, u32 lds_addr) {
    asm volatile("global_load_async_to_lds_b128 %0, %1, off"
                 :: "v"(lds_addr), "v"(gsrc) : "memory");
}

// ---------------------------------------------------------------------------
// Kernel 1: convert + transpose weights: W[H,C,DH] f32 -> Wt[H,DH,C] bf16
// ---------------------------------------------------------------------------
__global__ __launch_bounds__(256) void conv_w(const float* __restrict__ wq,
                                              const float* __restrict__ wk,
                                              const float* __restrict__ wv,
                                              u16* __restrict__ wt) {
    int which = blockIdx.z;
    const float* src = (which == 0) ? wq : (which == 1) ? wk : wv;
    u16* dst = wt + (size_t)which * Hn * DHn * Cn;
    int h = blockIdx.y;
    int tid = blockIdx.x * 256 + threadIdx.x;  // over DH*C = 65536
    int d = tid >> 10;
    int c = tid & 1023;
    dst[((size_t)h * DHn + d) * Cn + c] = f2bf(src[((size_t)h * Cn + c) * DHn + d]);
}

// ---------------------------------------------------------------------------
// Kernel 2: QKV projection GEMM with bf16 WMMA.
// Weights double-buffered in LDS via global_load_async_to_lds_b128 and shared
// by all 8 waves; x converted f32->bf16 in registers with hardware cvt.
//   q,k -> [B,H,T,DH] bf16 (row major);  v -> [B,H,DH,T] bf16 (transposed)
// ---------------------------------------------------------------------------
__global__ __launch_bounds__(256) void qkv_gemm(const float* __restrict__ x,
                                                const u16* __restrict__ wqt,
                                                const u16* __restrict__ wkt,
                                                const u16* __restrict__ wvt,
                                                u16* __restrict__ qo,
                                                u16* __restrict__ ko,
                                                u16* __restrict__ vto) {
    // [buf][w][dh-row 0..63][32 K + 8 pad]  (row stride 80B, 16B aligned)
    __shared__ alignas(16) u16 wstage[2][3][64][40];
    // per-wave output transpose staging (row stride 160B, 16B aligned)
    __shared__ alignas(16) u16 tlds[8][16][80];

    const int wave = threadIdx.x >> 5, lane = threadIdx.x & 31;
    const int m = lane & 15, half = lane >> 4;
    const int b = blockIdx.x >> 4;
    const int tblk = blockIdx.x & 15;
    const int h = blockIdx.y;
    const int tw = tblk * 128 + wave * 16;

    const u16* wptr[3] = {wqt, wkt, wvt};
    // cooperative async stage of the three 64x32 bf16 weight tiles at K=c0
    const int srow = threadIdx.x >> 2;        // 0..63
    const int schunk = (threadIdx.x & 3) * 8; // 0,8,16,24 (u16 units)
    auto issueW = [&](int c0, int buf) {
#pragma unroll
        for (int w = 0; w < 3; ++w) {
            const u16* g = wptr[w] + ((size_t)h * DHn + srow) * Cn + c0 + schunk;
            async_b128(g, (u32)(uintptr_t)&wstage[buf][w][srow][schunk]);
        }
    };

    v8f aq[4], ak[4], av[4];
#pragma unroll
    for (int n = 0; n < 4; ++n) {
        aq[n] = (v8f)(0.0f);
        ak[n] = (v8f)(0.0f);
        av[n] = (v8f)(0.0f);
    }

    const float* xrow = x + ((size_t)b * Tn + tw + m) * Cn;

    issueW(0, 0);
    for (int c0 = 0, it = 0; c0 < Cn; c0 += 32, ++it) {
        const int buf = it & 1;
        const bool hasNext = (c0 + 32 < Cn);
        if (hasNext) {
            issueW(c0 + 32, buf ^ 1);
            asm volatile("s_wait_asynccnt 0x3" ::: "memory");
        } else {
            asm volatile("s_wait_asynccnt 0x0" ::: "memory");
        }
        __syncthreads();  // current buffer visible to all waves

        // A fragment: 16x32 tile of x, f32 -> bf16 via hardware cvt.
        Frag a;
        const float* xp = xrow + c0 + half * 16;
        float4 f0 = *(const float4*)(xp);
        float4 f1 = *(const float4*)(xp + 4);
        float4 f2 = *(const float4*)(xp + 8);
        float4 f3 = *(const float4*)(xp + 12);
        a.b[0] = (__bf16)f0.x;  a.b[1] = (__bf16)f0.y;  a.b[2] = (__bf16)f0.z;  a.b[3] = (__bf16)f0.w;
        a.b[4] = (__bf16)f1.x;  a.b[5] = (__bf16)f1.y;  a.b[6] = (__bf16)f1.z;  a.b[7] = (__bf16)f1.w;
        a.b[8] = (__bf16)f2.x;  a.b[9] = (__bf16)f2.y;  a.b[10] = (__bf16)f2.z; a.b[11] = (__bf16)f2.w;
        a.b[12] = (__bf16)f3.x; a.b[13] = (__bf16)f3.y; a.b[14] = (__bf16)f3.z; a.b[15] = (__bf16)f3.w;

#pragma unroll
        for (int n = 0; n < 4; ++n) {
            const int row = n * 16 + m;
            Frag bq, bk, bv;
            const u16* wpq = &wstage[buf][0][row][half * 16];
            bq.q[0] = *(const uint4*)(wpq);
            bq.q[1] = *(const uint4*)(wpq + 8);
            aq[n] = __builtin_amdgcn_wmma_f32_16x16x32_bf16(false, a.v, false, bq.v,
                                                            (short)0, aq[n], false, false);
            const u16* wpk = &wstage[buf][1][row][half * 16];
            bk.q[0] = *(const uint4*)(wpk);
            bk.q[1] = *(const uint4*)(wpk + 8);
            ak[n] = __builtin_amdgcn_wmma_f32_16x16x32_bf16(false, a.v, false, bk.v,
                                                            (short)0, ak[n], false, false);
            const u16* wpv = &wstage[buf][2][row][half * 16];
            bv.q[0] = *(const uint4*)(wpv);
            bv.q[1] = *(const uint4*)(wpv + 8);
            av[n] = __builtin_amdgcn_wmma_f32_16x16x32_bf16(false, a.v, false, bv.v,
                                                            (short)0, av[n], false, false);
        }
        __syncthreads();  // all waves done with buf before it is overwritten
    }

    // v: store transposed [B,H,DH,T]. C-layout lane holds 8 consecutive t for one dh.
#pragma unroll
    for (int n = 0; n < 4; ++n) {
        uint4 pk;
        pk.x = f2bf(av[n][0]) | ((u32)f2bf(av[n][1]) << 16);
        pk.y = f2bf(av[n][2]) | ((u32)f2bf(av[n][3]) << 16);
        pk.z = f2bf(av[n][4]) | ((u32)f2bf(av[n][5]) << 16);
        pk.w = f2bf(av[n][6]) | ((u32)f2bf(av[n][7]) << 16);
        u16* vp = vto + ((size_t)(b * Hn + h) * DHn + n * 16 + m) * Tn + tw + half * 8;
        *(uint4*)vp = pk;
    }

    // q, k: transpose through per-wave LDS tile, then coalesced b128 stores.
#pragma unroll
    for (int o = 0; o < 2; ++o) {
        const v8f* acc = (o == 0) ? aq : ak;
        u16* dst = (o == 0) ? qo : ko;
#pragma unroll
        for (int n = 0; n < 4; ++n)
#pragma unroll
            for (int r = 0; r < 8; ++r)
                tlds[wave][r + 8 * half][n * 16 + m] = f2bf(acc[n][r]);
        asm volatile("" ::: "memory");  // same-wave DS ops are in-order in HW
        const int row = lane >> 1, part = lane & 1;
        const u16* lp = &tlds[wave][row][part * 32];
        uint4 d0 = *(const uint4*)(lp);
        uint4 d1 = *(const uint4*)(lp + 8);
        uint4 d2 = *(const uint4*)(lp + 16);
        uint4 d3 = *(const uint4*)(lp + 24);
        u16* gp = dst + ((size_t)(b * Hn + h) * Tn + tw + row) * DHn + part * 32;
        *(uint4*)(gp) = d0;
        *(uint4*)(gp + 8) = d1;
        *(uint4*)(gp + 16) = d2;
        *(uint4*)(gp + 24) = d3;
        asm volatile("" ::: "memory");
    }
}

// ---------------------------------------------------------------------------
// Kernel 3: causal flash attention. One wave = 16 query rows; key blocks of 32.
// ---------------------------------------------------------------------------
__global__ __launch_bounds__(256) void attn(const u16* __restrict__ q,
                                            const u16* __restrict__ k,
                                            const u16* __restrict__ vt,
                                            float* __restrict__ out) {
    __shared__ alignas(16) u16 plds[8][16][40];  // per-wave P tile, row stride 80B
    const int wave = threadIdx.x >> 5, lane = threadIdx.x & 31;
    const int m = lane & 15, half = lane >> 4;
    const int mblk = blockIdx.x;
    const int bh = blockIdx.y;
    const int b = bh >> 4, h = bh & 15;
    const int qrow0 = mblk * 128 + wave * 16;

    const u16* qbase = q + (size_t)bh * Tn * DHn;
    const u16* kbase = k + (size_t)bh * Tn * DHn;
    const u16* vbase = vt + (size_t)bh * DHn * Tn;

    // Preload q A-fragments (dh 0-31 and 32-63), direct contiguous 16B loads.
    Frag qa[2];
#pragma unroll
    for (int f = 0; f < 2; ++f) {
        const u16* qp = qbase + (size_t)(qrow0 + m) * DHn + f * 32 + half * 16;
        qa[f].q[0] = *(const uint4*)(qp);
        qa[f].q[1] = *(const uint4*)(qp + 8);
    }

    v8f acc[4];
    float m2[8], li[8];
#pragma unroll
    for (int n = 0; n < 4; ++n) acc[n] = (v8f)(0.0f);
#pragma unroll
    for (int r = 0; r < 8; ++r) { m2[r] = -1.0e30f; li[r] = 0.0f; }

    const float kScale = 0.125f * 1.44269504088896340736f;  // 1/sqrt(DH) * log2(e)

    for (int s0 = 0; s0 < qrow0 + 16; s0 += 32) {
        // Prefetch next key/value block while this one computes.
        if (s0 + 32 < qrow0 + 16) {
            __builtin_prefetch(kbase + (size_t)(s0 + 32 + lane) * DHn, 0, 0);
            __builtin_prefetch(vbase + (size_t)lane * Tn + s0 + 32, 0, 0);
            __builtin_prefetch(vbase + (size_t)(lane + 32) * Tn + s0 + 32, 0, 0);
        }

        // ----- S = q @ k^T for this 16x32 key block (two 16x16 f32 tiles) -----
        v8f s[2];
#pragma unroll
        for (int t = 0; t < 2; ++t) {
            const u16* kp = kbase + (size_t)(s0 + t * 16 + m) * DHn + half * 16;
            Frag k0, k1;
            k0.q[0] = *(const uint4*)(kp);
            k0.q[1] = *(const uint4*)(kp + 8);
            k1.q[0] = *(const uint4*)(kp + 32);
            k1.q[1] = *(const uint4*)(kp + 40);
            v8f z = (v8f)(0.0f);
            z = __builtin_amdgcn_wmma_f32_16x16x32_bf16(false, qa[0].v, false, k0.v,
                                                        (short)0, z, false, false);
            z = __builtin_amdgcn_wmma_f32_16x16x32_bf16(false, qa[1].v, false, k1.v,
                                                        (short)0, z, false, false);
            s[t] = z;
        }

        // ----- online softmax (base-2), causal mask, write P(bf16) to LDS -----
#pragma unroll
        for (int r = 0; r < 8; ++r) {
            const int qr = qrow0 + r + 8 * half;
            float a0 = s[0][r] * kScale;
            if (s0 + m > qr) a0 = -1.0e30f;
            float a1 = s[1][r] * kScale;
            if (s0 + 16 + m > qr) a1 = -1.0e30f;
            float mx = red_max16(fmaxf(a0, a1));
            float mn = fmaxf(m2[r], mx);
            float al = fexp2(m2[r] - mn);
            float p0 = fexp2(a0 - mn);
            float p1 = fexp2(a1 - mn);
            float ps = red_sum16(p0 + p1);
            li[r] = li[r] * al + ps;
            m2[r] = mn;
            acc[0][r] *= al;
            acc[1][r] *= al;
            acc[2][r] *= al;
            acc[3][r] *= al;
            const int row = r + 8 * half;
            plds[wave][row][m] = f2bf(p0);
            plds[wave][row][16 + m] = f2bf(p1);
        }
        asm volatile("" ::: "memory");

        // ----- read P back in A-fragment layout, O += P @ V -----
        Frag pa;
        pa.q[0] = *(const uint4*)&plds[wave][m][half * 16];
        pa.q[1] = *(const uint4*)&plds[wave][m][half * 16 + 8];
#pragma unroll
        for (int n = 0; n < 4; ++n) {
            const u16* vp = vbase + (size_t)(n * 16 + m) * Tn + s0 + half * 16;
            Frag vf;
            vf.q[0] = *(const uint4*)(vp);
            vf.q[1] = *(const uint4*)(vp + 8);
            acc[n] = __builtin_amdgcn_wmma_f32_16x16x32_bf16(false, pa.v, false, vf.v,
                                                             (short)0, acc[n], false, false);
        }
        asm volatile("" ::: "memory");
    }

    // ----- epilogue: O / l, write to out[B,T,H*DH] fp32 -----
#pragma unroll
    for (int r = 0; r < 8; ++r) {
        const float inv = 1.0f / li[r];
        const int row = qrow0 + r + 8 * half;
        float* op = out + ((size_t)b * Tn + row) * (Hn * DHn) + h * DHn;
#pragma unroll
        for (int n = 0; n < 4; ++n) op[n * 16 + m] = acc[n][r] * inv;
    }
}

// ---------------------------------------------------------------------------
extern "C" void kernel_launch(void* const* d_in, const int* in_sizes, int n_in,
                              void* d_out, int out_size, void* d_ws, size_t ws_size,
                              hipStream_t stream) {
    const float* x = (const float*)d_in[0];
    const float* Wq = (const float*)d_in[1];
    const float* Wk = (const float*)d_in[2];
    const float* Wv = (const float*)d_in[3];
    float* out = (float*)d_out;

    u16* ws = (u16*)d_ws;
    const size_t WT = (size_t)Hn * DHn * Cn;      // per-weight bf16 elements
    const size_t QE = (size_t)Bn * Hn * Tn * DHn; // q/k/v bf16 elements
    u16* wqt = ws;
    u16* wkt = ws + WT;
    u16* wvt = ws + 2 * WT;
    u16* qb = ws + 3 * WT;
    u16* kb = qb + QE;
    u16* vtb = kb + QE;
    // total workspace: (3*WT + 3*QE)*2 bytes ~= 56.6 MB

    conv_w<<<dim3((DHn * Cn) / 256, Hn, 3), 256, 0, stream>>>(Wq, Wk, Wv, ws);
    qkv_gemm<<<dim3(Bn * (Tn / 128), Hn), 256, 0, stream>>>(x, wqt, wkt, wvt, qb, kb, vtb);
    attn<<<dim3(Tn / 128, Bn * Hn), 256, 0, stream>>>(qb, kb, vtb, out);
}
